// LatticeSnake_37598143709498
// MI455X (gfx1250) — matrix operand order repaired and checked by reading, not compiled, not run on hardware
//
#include <hip/hip_runtime.h>

// CDNA5 / gfx1250: wave32, WMMA 16x16x32 f16 -> f32
typedef __attribute__((ext_vector_type(16))) _Float16 v16h;
typedef __attribute__((ext_vector_type(8)))  float    v8f;

#define BATCH   16
#define NPT     48      // N in reference
#define KW      7       // window size
#define PADW    3
#define NSLOT   128     // padded point table: [0,48) acids, [64,111) inter
#define NCOL    49      // 7*7 (j,k) columns
#define OUTPER  343     // 7*7*7 per (b,n)

union HalfPack { _Float16 h[2]; unsigned u; };

__global__ __launch_bounds__(256)
void lattice_snake_wmma(const float* __restrict__ acids,
                        const unsigned char* __restrict__ mask,
                        const int* __restrict__ idx,
                        float* __restrict__ out)
{
    __shared__ unsigned pts_c[NSLOT];   // packed c0|c1<<8|c2<<16|valid<<24
    __shared__ unsigned pts_v[NSLOT];   // packed {f16 hi, f16 lo} of value

    const int b    = blockIdx.x / 6;    // batch: 8 queries per block, 6 blocks/batch
    const int ngrp = blockIdx.x % 6;
    const int tid  = threadIdx.x;

    // ---- build the per-batch point table in LDS (95 live points) ----
    if (tid < NSLOT) {
        int c0 = 0, c1 = 0, c2 = 0, valid = 0; float v = 0.0f;
        if (tid < NPT) {                      // acid points
            int p = tid;
            c0 = 2 * (idx[(b*NPT + p)*3 + 0] + (NPT - 1)) + PADW;
            c1 = 2 * (idx[(b*NPT + p)*3 + 1] + (NPT - 1)) + PADW;
            c2 = 2 * (idx[(b*NPT + p)*3 + 2] + (NPT - 1)) + PADW;
            valid = mask[b*NPT + p] ? 1 : 0;
            v = acids[b*NPT + p];
        } else if (tid >= 64 && tid < 64 + (NPT - 1)) {   // inter points
            int q = tid - 64;
            int a0 = 2 * (idx[(b*NPT + q    )*3 + 0] + (NPT - 1));
            int a1 = 2 * (idx[(b*NPT + q    )*3 + 1] + (NPT - 1));
            int a2 = 2 * (idx[(b*NPT + q    )*3 + 2] + (NPT - 1));
            int e0 = 2 * (idx[(b*NPT + q + 1)*3 + 0] + (NPT - 1));
            int e1 = 2 * (idx[(b*NPT + q + 1)*3 + 1] + (NPT - 1));
            int e2 = 2 * (idx[(b*NPT + q + 1)*3 + 2] + (NPT - 1));
            c0 = ((a0 + e0) >> 1) + PADW;
            c1 = ((a1 + e1) >> 1) + PADW;
            c2 = ((a2 + e2) >> 1) + PADW;
            valid = mask[b*NPT + q + 1] ? 1 : 0;
            v = acids[b*NPT + q] + acids[b*NPT + q + 1] + 1.0f;
        }
        _Float16 hi = (_Float16)v;
        _Float16 lo = (_Float16)(v - (float)hi);   // hi+lo == v to ~f32 precision
        HalfPack hp; hp.h[0] = hi; hp.h[1] = lo;
        pts_c[tid] = (unsigned)(c0 & 0xff) | ((unsigned)(c1 & 0xff) << 8) |
                     ((unsigned)(c2 & 0xff) << 16) | ((unsigned)valid << 24);
        pts_v[tid] = hp.u;
    }
    __syncthreads();

    const int wave = tid >> 5;
    const int lane = tid & 31;
    const int half = lane >> 4;       // lane half (wave32 WMMA layout)
    const int m    = lane & 15;       // A row or B/C column within tile
    const int n    = ngrp * 8 + wave; // query index

    const int q0 = 2 * (idx[(b*NPT + n)*3 + 0] + (NPT - 1));
    const int q1 = 2 * (idx[(b*NPT + n)*3 + 1] + (NPT - 1));
    const int q2 = 2 * (idx[(b*NPT + n)*3 + 2] + (NPT - 1));
    const float maskf = mask[b*NPT + n] ? 1.0f : 0.0f;

    // ---- A fragments (16x32 f16 per k-step):
    //  rows 0..6  : Ex[p,i] * value (hi pass / lo pass)
    //  rows 8..14 : Ex[p,i] * 1  (match counter, hi pass only)
    v16h a_hi[4], a_lo[4];
#pragma unroll
    for (int kt = 0; kt < 4; ++kt) {
#pragma unroll
        for (int e = 0; e < 16; ++e) {
            // 16-bit A layout: elem e, lane-half h -> K = (e<8 ? h*8+e : 16+h*8+e-8)
            int klocal = (e < 8) ? (half*8 + e) : (16 + half*8 + (e - 8));
            int slot   = kt*32 + klocal;
            unsigned pc = pts_c[slot];
            unsigned pv = pts_v[slot];
            int d0    = (int)(pc & 0xff) - q0;
            int valid = (int)((pc >> 24) & 1);
            HalfPack hp; hp.u = pv;
            _Float16 ah = (_Float16)0.0f, al = (_Float16)0.0f;
            if (valid) {
                if (m < 7 && d0 == m)                       { ah = hp.h[0]; al = hp.h[1]; }
                else if (m >= 8 && m < 15 && d0 == (m - 8)) { ah = (_Float16)1.0f; }
            }
            a_hi[kt][e] = ah;
            a_lo[kt][e] = al;
        }
    }

    float* outp = out + (size_t)(b*NPT + n) * OUTPER;

#pragma unroll
    for (int t = 0; t < 4; ++t) {          // 4 tiles of 16 columns (64 >= 49)
        v8f c_a = {};                      // acids accumulator
        v8f c_i = {};                      // inter accumulator (+ counts rows 8..14)
#pragma unroll
        for (int kt = 0; kt < 4; ++kt) {
            // B fragment (32x16 f16): elem e, lane-half h -> K = (e<8 ? h*8+e : 16+h*8+e-8), column = m
            v16h bb;
#pragma unroll
            for (int e = 0; e < 16; ++e) {
                int slot = kt*32 + ((e < 8) ? (half*8 + e) : (16 + half*8 + (e - 8)));
                unsigned pc = pts_c[slot];
                int valid = (int)((pc >> 24) & 1);
                int d1 = (int)((pc >>  8) & 0xff) - q1;
                int d2 = (int)((pc >> 16) & 0xff) - q2;
                int lin = (valid && (unsigned)d1 < 7u && (unsigned)d2 < 7u) ? (d1*7 + d2) : -1;
                bb[e] = (lin == t*16 + m) ? (_Float16)1.0f : (_Float16)0.0f;
            }
            if (kt < 2) {  // k-slots [0,64): acids (+padding)
                c_a = __builtin_amdgcn_wmma_f32_16x16x32_f16(false, a_hi[kt], false, bb, (short)0, c_a, false, false);
                c_a = __builtin_amdgcn_wmma_f32_16x16x32_f16(false, a_lo[kt], false, bb, (short)0, c_a, false, false);
            } else {       // k-slots [64,128): inter (+padding)
                c_i = __builtin_amdgcn_wmma_f32_16x16x32_f16(false, a_hi[kt], false, bb, (short)0, c_i, false, false);
                c_i = __builtin_amdgcn_wmma_f32_16x16x32_f16(false, a_lo[kt], false, bb, (short)0, c_i, false, false);
            }
        }
        // C/D layout: elem v on lane (h,m) = row v+8h, col m. Value rows live on h=0,
        // matching count row (v+8) lives on the h=1 partner lane -> shfl_xor 16.
#pragma unroll
        for (int v = 0; v < 7; ++v) {
            float cnt = __shfl_xor(c_i[v], 16, 32);   // inter match count for (row v, col m)
            float vi  = c_i[v];
            float va  = c_a[v];
            int col = t*16 + m;
            if (half == 0 && col < NCOL) {
                float r = (cnt > 0.5f) ? vi : va;     // inter overwrites acids
                outp[v*NCOL + col] = r * maskf;       // i*49 + j*7 + k
            }
        }
    }
}

extern "C" void kernel_launch(void* const* d_in, const int* in_sizes, int n_in,
                              void* d_out, int out_size, void* d_ws, size_t ws_size,
                              hipStream_t stream) {
    const float*         acids = (const float*)d_in[0];          // (16,48) f32
    const unsigned char* mask  = (const unsigned char*)d_in[1];  // (16,48) bool (1 byte)
    const int*           idx   = (const int*)d_in[2];            // (16,48,3) i32
    float*               out   = (float*)d_out;                  // (16,48,7,7,7,1) f32

    // 96 blocks x 256 threads: block = (batch, group of 8 queries); 1 wave per query.
    lattice_snake_wmma<<<dim3(BATCH * 6), dim3(256), 0, stream>>>(acids, mask, idx, out);
}
